// CrossAttention_80599356277365
// MI455X (gfx1250) — matrix operand order repaired and checked by reading, not compiled, not run on hardware
//
#include <hip/hip_runtime.h>
#include <hip/hip_bf16.h>

// ---------------------------------------------------------------------------
// CrossAttention (GroupNorm -> QKV proj -> 8-head attn -> out proj + residual)
// MI455X / gfx1250: wave32, WMMA f32_16x16x32_f16 for all GEMM-shaped work.
// ---------------------------------------------------------------------------

typedef __attribute__((ext_vector_type(16))) _Float16 v16h;
typedef __attribute__((ext_vector_type(8)))  _Float16 v8h;
typedef __attribute__((ext_vector_type(4)))  _Float16 v4h;
typedef __attribute__((ext_vector_type(8)))  float    v8f;

union F16x16 { v16h v; v8h h[2]; };

#define B_   8
#define HW_  4096      // 64*64
#define C_   512
#define S_   256
#define CD_  768
#define NH_  8
#define HD_  64
#define NG_  32
#define EPS_ 1e-5f

__device__ __forceinline__ v8f wmma_f16(v16h a, v16h b, v8f c) {
  // D = A(16x32 f16) * B(32x16 f16) + C(16x16 f32)
  return __builtin_amdgcn_wmma_f32_16x16x32_f16(
      /*neg_a=*/false, a, /*neg_b=*/false, b,
      /*c_mod=*/(short)0, c, /*reuse_a=*/false, /*reuse_b=*/false);
}

// A row-major [M,K] f16. Fragment rows m0..m0+15 (row passed per-lane), k0..k0+31.
// A k order used here: lanes 0-15 K=k0..k0+15 ; lanes 16-31 K=k0+16..k0+31.
__device__ __forceinline__ v16h load_a(const _Float16* __restrict__ A, int ldk,
                                       int row, int k0, int lane) {
  const _Float16* p = A + (size_t)row * ldk + k0 + ((lane >> 4) << 4);
  F16x16 f;
  f.h[0] = *(const v8h*)p;
  f.h[1] = *(const v8h*)(p + 8);
  return f.v;
}

// Bt row-major [N,K] f16 (B pre-transposed). Fragment col n (per-lane), k0..k0+31.
// B layout: lanes 0-15 hold K=k0..k0+15, lanes 16-31 hold K=k0+16..k0+31.
__device__ __forceinline__ v16h load_bt(const _Float16* __restrict__ Bt, int ldk,
                                        int col, int k0, int lane) {
  const _Float16* p = Bt + (size_t)col * ldk + k0 + ((lane >> 4) << 4);
  F16x16 f;
  f.h[0] = *(const v8h*)p;
  f.h[1] = *(const v8h*)(p + 8);
  return f.v;
}

// ---------------------------------------------------------------------------
// Prep kernels
// ---------------------------------------------------------------------------

// W[K][N] fp32  ->  Wt[N][K] f16
__global__ void wtrans_kernel(const float* __restrict__ W, _Float16* __restrict__ Wt,
                              int K, int N) {
  int i = blockIdx.x * blockDim.x + threadIdx.x;
  if (i >= K * N) return;
  int k = i / N, n = i - k * N;
  Wt[(size_t)n * K + k] = (_Float16)W[i];
}

// fp32 -> f16, 4 elements per thread
__global__ void cvt16_kernel(const float* __restrict__ in, _Float16* __restrict__ out,
                             int n4) {
  int i = blockIdx.x * blockDim.x + threadIdx.x;
  if (i >= n4) return;
  float4 v = ((const float4*)in)[i];
  v4h o = { (_Float16)v.x, (_Float16)v.y, (_Float16)v.z, (_Float16)v.w };
  ((v4h*)out)[i] = o;
}

// GroupNorm stats: one block per (b,g); reduce 64*64*16 = 65536 values.
__global__ __launch_bounds__(256) void gn_stats_kernel(const float* __restrict__ x,
                                                       float* __restrict__ stats) {
  int bg = blockIdx.x;
  int b = bg >> 5, g = bg & 31;
  const float* xb = x + (size_t)b * HW_ * C_ + g * 16;
  float s = 0.f, s2 = 0.f;
  for (int i = threadIdx.x; i < HW_ * 16; i += 256) {
    int hw = i >> 4, c = i & 15;
    float v = xb[(size_t)hw * C_ + c];
    s += v; s2 += v * v;
  }
  __shared__ float sh[512];
  sh[threadIdx.x] = s;
  sh[256 + threadIdx.x] = s2;
  __syncthreads();
  for (int off = 128; off > 0; off >>= 1) {
    if ((int)threadIdx.x < off) {
      sh[threadIdx.x]       += sh[threadIdx.x + off];
      sh[256 + threadIdx.x] += sh[256 + threadIdx.x + off];
    }
    __syncthreads();
  }
  if (threadIdx.x == 0) {
    const float inv_n = 1.0f / (HW_ * 16);
    float mean = sh[0] * inv_n;
    float var  = sh[256] * inv_n - mean * mean;
    stats[bg * 2 + 0] = mean;
    stats[bg * 2 + 1] = rsqrtf(var + EPS_);
  }
}

// Normalize + affine + f16 convert (4 channels per thread, same group).
__global__ void gn_apply_kernel(const float* __restrict__ x,
                                const float* __restrict__ stats,
                                const float* __restrict__ scale,
                                const float* __restrict__ bias,
                                _Float16* __restrict__ xn) {
  int i = blockIdx.x * blockDim.x + threadIdx.x;   // element-quad index
  size_t base = (size_t)i * 4;
  int c = (int)(base & (C_ - 1));
  int b = (int)(base >> 21);                        // HW_*C_ = 2^21
  int g = c >> 4;
  float mean = stats[(b * NG_ + g) * 2 + 0];
  float rstd = stats[(b * NG_ + g) * 2 + 1];
  float4 v = ((const float4*)x)[i];
  v4h o;
  o.x = (_Float16)((v.x - mean) * rstd * scale[c + 0] + bias[c + 0]);
  o.y = (_Float16)((v.y - mean) * rstd * scale[c + 1] + bias[c + 1]);
  o.z = (_Float16)((v.z - mean) * rstd * scale[c + 2] + bias[c + 2]);
  o.w = (_Float16)((v.w - mean) * rstd * scale[c + 3] + bias[c + 3]);
  ((v4h*)xn)[i] = o;
}

// v[b][t][c] -> vT[b][c][t]  (f16)
__global__ void vtrans_kernel(const _Float16* __restrict__ V, _Float16* __restrict__ VT) {
  int i = blockIdx.x * blockDim.x + threadIdx.x;    // output index over B_*C_*S_
  int b = i >> 17;                                  // C_*S_ = 2^17
  int rem = i & ((1 << 17) - 1);
  int c = rem >> 8;
  int t = rem & (S_ - 1);
  VT[i] = V[((size_t)(b * S_ + t)) * C_ + c];
}

// ---------------------------------------------------------------------------
// WMMA GEMM: C[M,N] = A[M,K] @ Bt[N,K]^T.
// Block = 8 waves, wave tile 32x64 (2 A-frags share every B-frag: 12 b128
// loads per 8 WMMAs). Weights (Bt) stay L2-resident; A streams with prefetch.
// FINAL==1: fp32 out with +bias[col] +resid[row,col] (residual add epilogue).
// ---------------------------------------------------------------------------
template <int FINAL>
__global__ __launch_bounds__(256) void gemm_kernel(const _Float16* __restrict__ A,
                                                   const _Float16* __restrict__ Bt,
                                                   int M, int N, int K,
                                                   _Float16* __restrict__ Of16,
                                                   float* __restrict__ Of32,
                                                   const float* __restrict__ bias,
                                                   const float* __restrict__ resid) {
  int lane = threadIdx.x;                   // 0..31
  int wv   = threadIdx.y;                   // 0..7
  int li = lane & 15, lh = lane >> 4;
  int m0 = (blockIdx.x * 8 + wv) * 32;
  int n0 = blockIdx.y * 64;
  int row0 = m0 + li;
  int row1 = m0 + 16 + li;

  v8f acc0[4] = {};
  v8f acc1[4] = {};
  for (int k0 = 0; k0 < K; k0 += 32) {
    v16h a0 = load_a(A, K, row0, k0, lane);
    v16h a1 = load_a(A, K, row1, k0, lane);
    if (k0 + 64 < K) {
      // A is the streaming operand (activations); pull next-next step line.
      __builtin_prefetch((const void*)(A + (size_t)row0 * K + k0 + 64), 0, 0);
      __builtin_prefetch((const void*)(A + (size_t)row1 * K + k0 + 64), 0, 0);
    }
#pragma unroll
    for (int j = 0; j < 4; ++j) {
      v16h b = load_bt(Bt, K, n0 + j * 16 + li, k0, lane);
      acc0[j] = wmma_f16(a0, b, acc0[j]);
      acc1[j] = wmma_f16(a1, b, acc1[j]);
    }
  }

#pragma unroll
  for (int half = 0; half < 2; ++half) {
    int rb = m0 + half * 16 + lh * 8;       // C layout: lanes 16-31 rows +8
    v8f* acc = half ? acc1 : acc0;
#pragma unroll
    for (int j = 0; j < 4; ++j) {
      int cc = n0 + j * 16 + li;
#pragma unroll
      for (int r = 0; r < 8; ++r) {
        size_t idx = (size_t)(rb + r) * N + cc;
        if (FINAL) {
          Of32[idx] = acc[j][r] + bias[cc] + resid[idx];
        } else {
          Of16[idx] = (_Float16)acc[j][r];
        }
      }
    }
  }
}

// ---------------------------------------------------------------------------
// Attention: grid (4096/64, NH_, B_), block (32,4). One wave = 16 queries,
// full 256-key softmax held in registers (16 score tiles).
// ---------------------------------------------------------------------------
#define LST 264   // padded LDS row stride in halfs (16B aligned, conflict-free)

__global__ __launch_bounds__(128) void attn_kernel(const _Float16* __restrict__ Q,
                                                   const _Float16* __restrict__ Km,
                                                   const _Float16* __restrict__ VT,
                                                   _Float16* __restrict__ O) {
  __shared__ __align__(16) _Float16 plds[4 * 16 * LST];

  int lane = threadIdx.x, wv = threadIdx.y;
  int li = lane & 15, lh = lane >> 4;
  int b = blockIdx.z, h = blockIdx.y;
  int q0 = blockIdx.x * 64 + wv * 16;

  const _Float16* Qb = Q  + (size_t)b * HW_ * C_ + h * HD_;
  const _Float16* Kb = Km + (size_t)b * S_  * C_ + h * HD_;
  const _Float16* Vb = VT + ((size_t)b * C_ + h * HD_) * S_;

  // Q fragments for the two 32-wide steps over head_dim (64).
  v16h qf[2];
  {
    const _Float16* qrow = Qb + (size_t)(q0 + li) * C_;
#pragma unroll
    for (int ks = 0; ks < 2; ++ks) {
      const _Float16* p = qrow + ks * 32 + (lh << 4);
      F16x16 f; f.h[0] = *(const v8h*)p; f.h[1] = *(const v8h*)(p + 8);
      qf[ks] = f.v;
    }
  }

  // Scores: 16 tiles of 16x16 (q x key) = full (16 x 256) strip.
  v8f sc[16] = {};
#pragma unroll
  for (int ks = 0; ks < 2; ++ks) {
#pragma unroll
    for (int nt = 0; nt < 16; ++nt) {
      const _Float16* kr = Kb + (size_t)(nt * 16 + li) * C_ + ks * 32 + (lh << 4);
      F16x16 f; f.h[0] = *(const v8h*)kr; f.h[1] = *(const v8h*)(kr + 8);
      sc[nt] = wmma_f16(qf[ks], f.v, sc[nt]);
    }
  }

  // Softmax over 256 keys. Row (q) for VGPR slot r is (r + 8*lh); the 16
  // columns live across lanes of the same half -> butterfly over masks 1..8.
  float rmax[8], rinv[8];
#pragma unroll
  for (int r = 0; r < 8; ++r) {
    float m = sc[0][r];
#pragma unroll
    for (int nt = 1; nt < 16; ++nt) m = fmaxf(m, sc[nt][r]);
    m = fmaxf(m, __shfl_xor(m, 1, 32));
    m = fmaxf(m, __shfl_xor(m, 2, 32));
    m = fmaxf(m, __shfl_xor(m, 4, 32));
    m = fmaxf(m, __shfl_xor(m, 8, 32));
    rmax[r] = m;
  }
  const float isc = 0.125f;   // 1/sqrt(64); softmax(s/8) = exp((s-max)/8)/sum
#pragma unroll
  for (int r = 0; r < 8; ++r) {
    float s = 0.f;
#pragma unroll
    for (int nt = 0; nt < 16; ++nt) {
      float e = __expf((sc[nt][r] - rmax[r]) * isc);
      sc[nt][r] = e;
      s += e;
    }
    s += __shfl_xor(s, 1, 32);
    s += __shfl_xor(s, 2, 32);
    s += __shfl_xor(s, 4, 32);
    s += __shfl_xor(s, 8, 32);
    rinv[r] = 1.0f / s;
  }

  // Spill P (16x256) to LDS as f16 in row-major A-fragment-friendly layout.
  _Float16* pl = &plds[wv * 16 * LST];
#pragma unroll
  for (int nt = 0; nt < 16; ++nt) {
#pragma unroll
    for (int r = 0; r < 8; ++r) {
      pl[(r + 8 * lh) * LST + nt * 16 + li] = (_Float16)sc[nt][r];
    }
  }

  // out(16x64) = P(16x256) @ V(256x64), V pre-transposed so B-frags contiguous.
  v8f out[4] = {};
  for (int ts = 0; ts < 8; ++ts) {
    int t0 = ts * 32;
    F16x16 a;
    const _Float16* pr = pl + li * LST + t0 + (lh << 4);
    a.h[0] = *(const v8h*)pr;
    a.h[1] = *(const v8h*)(pr + 8);
#pragma unroll
    for (int j = 0; j < 4; ++j) {
      const _Float16* vr = Vb + (size_t)(j * 16 + li) * S_ + t0 + (lh << 4);
      F16x16 f; f.h[0] = *(const v8h*)vr; f.h[1] = *(const v8h*)(vr + 8);
      out[j] = wmma_f16(a.v, f.v, out[j]);
    }
  }

  // Normalize by row sums and write f16 attention output (b, q, h*64+d).
#pragma unroll
  for (int j = 0; j < 4; ++j) {
    int col = h * HD_ + j * 16 + li;
#pragma unroll
    for (int r = 0; r < 8; ++r) {
      int rowq = q0 + r + 8 * lh;
      O[((size_t)(b * HW_ + rowq)) * C_ + col] = (_Float16)(out[j][r] * rinv[r]);
    }
  }
}

// ---------------------------------------------------------------------------
// Host-side launch
// ---------------------------------------------------------------------------
extern "C" void kernel_launch(void* const* d_in, const int* in_sizes, int n_in,
                              void* d_out, int out_size, void* d_ws, size_t ws_size,
                              hipStream_t stream) {
  (void)in_sizes; (void)n_in; (void)out_size; (void)ws_size;
  const float* x    = (const float*)d_in[0];   // (8,64,64,512)
  const float* cond = (const float*)d_in[1];   // (8,256,768)
  const float* gns  = (const float*)d_in[2];   // (512,)
  const float* gnb  = (const float*)d_in[3];   // (512,)
  const float* Wq   = (const float*)d_in[4];   // (512,512)
  const float* Wk   = (const float*)d_in[5];   // (768,512)
  const float* Wv   = (const float*)d_in[6];   // (768,512)
  const float* Wo   = (const float*)d_in[7];   // (512,512)
  const float* bo   = (const float*)d_in[8];   // (512,)
  float* out = (float*)d_out;

  char* ws = (char*)d_ws;
  size_t off = 0;
  auto alloc = [&](size_t bytes) -> char* {
    char* p = ws + off;
    off += (bytes + 255) & ~(size_t)255;
    return p;
  };

  const size_t Mx = (size_t)B_ * HW_;   // 32768
  const size_t Mc = (size_t)B_ * S_;    // 2048

  float*     stats  = (float*)    alloc((size_t)B_ * NG_ * 2 * sizeof(float));
  _Float16*  xn     = (_Float16*) alloc(Mx * C_ * 2);
  _Float16*  cond16 = (_Float16*) alloc(Mc * CD_ * 2);
  _Float16*  WqT    = (_Float16*) alloc((size_t)C_ * C_ * 2);
  _Float16*  WkT    = (_Float16*) alloc((size_t)CD_ * C_ * 2);
  _Float16*  WvT    = (_Float16*) alloc((size_t)CD_ * C_ * 2);
  _Float16*  WoT    = (_Float16*) alloc((size_t)C_ * C_ * 2);
  _Float16*  q16    = (_Float16*) alloc(Mx * C_ * 2);
  _Float16*  k16    = (_Float16*) alloc(Mc * C_ * 2);
  _Float16*  v16    = (_Float16*) alloc(Mc * C_ * 2);
  _Float16*  vT     = (_Float16*) alloc(Mc * C_ * 2);
  _Float16*  ao     = (_Float16*) alloc(Mx * C_ * 2);

  // Weight prep (transpose + f16)
  wtrans_kernel<<<(C_ * C_) / 256, 256, 0, stream>>>(Wq, WqT, C_, C_);
  wtrans_kernel<<<(CD_ * C_) / 256, 256, 0, stream>>>(Wk, WkT, CD_, C_);
  wtrans_kernel<<<(CD_ * C_) / 256, 256, 0, stream>>>(Wv, WvT, CD_, C_);
  wtrans_kernel<<<(C_ * C_) / 256, 256, 0, stream>>>(Wo, WoT, C_, C_);

  // cond -> f16
  cvt16_kernel<<<(int)((Mc * CD_ / 4) / 256), 256, 0, stream>>>(cond, cond16,
                                                                (int)(Mc * CD_ / 4));
  // GroupNorm
  gn_stats_kernel<<<B_ * NG_, 256, 0, stream>>>(x, stats);
  gn_apply_kernel<<<(int)((Mx * C_ / 4) / 256), 256, 0, stream>>>(x, stats, gns, gnb, xn);

  // Projections (WMMA GEMMs); wave tile 32x64, block tile 256x64.
  dim3 blk(32, 8);
  gemm_kernel<0><<<dim3((unsigned)(Mc / 256), C_ / 64), blk, 0, stream>>>(
      cond16, WkT, (int)Mc, C_, CD_, k16, nullptr, nullptr, nullptr);
  gemm_kernel<0><<<dim3((unsigned)(Mc / 256), C_ / 64), blk, 0, stream>>>(
      cond16, WvT, (int)Mc, C_, CD_, v16, nullptr, nullptr, nullptr);
  gemm_kernel<0><<<dim3((unsigned)(Mx / 256), C_ / 64), blk, 0, stream>>>(
      xn, WqT, (int)Mx, C_, C_, q16, nullptr, nullptr, nullptr);

  // V transpose for PV WMMA
  vtrans_kernel<<<(int)((Mc * C_) / 256), 256, 0, stream>>>(v16, vT);

  // Attention
  attn_kernel<<<dim3(HW_ / 64, NH_, B_), dim3(32, 4), 0, stream>>>(q16, k16, vT, ao);

  // Output projection + bias + residual -> fp32 result
  gemm_kernel<1><<<dim3((unsigned)(Mx / 256), C_ / 64), blk, 0, stream>>>(
      ao, WoT, (int)Mx, C_, C_, nullptr, out, bo, x);
}
